// MultiHeadAttention_10075993277138
// MI455X (gfx1250) — compile-verified, matrix operand order checked by reading, not run on hardware
//
#include <hip/hip_runtime.h>

// ---------------- CDNA5 WMMA types ----------------
typedef __attribute__((ext_vector_type(16))) __bf16 v16bf;
typedef __attribute__((ext_vector_type(8)))  __bf16 v8bf;
typedef __attribute__((ext_vector_type(8)))  float  v8f;

// Problem constants (match reference)
#define BB 4
#define SS 2048
#define DD 1024
#define HH 16
#define DHH 64

#define WMMA_BF16(a, b, c) \
  __builtin_amdgcn_wmma_f32_16x16x32_bf16(false, (a), false, (b), (short)0, (c), false, false)

// ---------------- fp32 -> bf16 convert (vectorized) ----------------
__global__ void cvt_bf16_kernel(const float* __restrict__ in,
                                __bf16* __restrict__ out, int n) {
  int i = (blockIdx.x * blockDim.x + threadIdx.x) * 8;
  if (i + 7 < n) {
    float4 f0 = *(const float4*)(in + i);
    float4 f1 = *(const float4*)(in + i + 4);
    v8bf pk;
    pk[0] = (__bf16)f0.x; pk[1] = (__bf16)f0.y;
    pk[2] = (__bf16)f0.z; pk[3] = (__bf16)f0.w;
    pk[4] = (__bf16)f1.x; pk[5] = (__bf16)f1.y;
    pk[6] = (__bf16)f1.z; pk[7] = (__bf16)f1.w;
    *(v8bf*)(out + i) = pk;
  }
}

// ---------------- weight transpose + convert: Wt[n][k] = (bf16)W[k][n] ----
__global__ void transpose_w_kernel(const float* __restrict__ W,
                                   __bf16* __restrict__ Wt) {
  int idx = blockIdx.x * blockDim.x + threadIdx.x;   // over D*D
  int k = idx >> 10;
  int n = idx & (DD - 1);
  Wt[n * DD + k] = (__bf16)W[idx];
}

// ---------------- bf16 WMMA GEMM: C[M,N] = cscale * A[M,K] * Bt[N,K]^T ----
// Register-blocked: one wave computes a 32x64 tile (8 C fragments).
// Per k-step (K+=32): 2 A-frag loads + 4 B-frag loads -> 8 WMMAs.
// MODE 0: store bf16 head-split [B,H,S,DH]
// MODE 1: store bf16 head-split transposed [B,H,DH,S]   (for V)
// MODE 2: store fp32 [M,N]                              (final output)
template <int MODE>
__global__ void gemm_bf16_kernel(const __bf16* __restrict__ A,
                                 const __bf16* __restrict__ Bt,
                                 void* __restrict__ Out, float cscale) {
  const int K = DD, N = DD;
  int wave = threadIdx.x >> 5;
  int lane = threadIdx.x & 31;
  int wy = wave >> 2, wx = wave & 3;           // 2x4 wave grid in block
  int m0 = blockIdx.y * 64  + wy * 32;
  int n0 = blockIdx.x * 256 + wx * 64;
  int nl = lane & 15, half = lane >> 4;

  const __bf16* a0 = A  + (size_t)(m0 + nl) * K + half * 16;   // A frag: lane=row
  const __bf16* a1 = a0 + (size_t)16 * K;
  const __bf16* b0 = Bt + (size_t)(n0 + nl) * K + half * 16;   // B frag: lane=col

  v8f c[2][4] = {};
#pragma unroll 2
  for (int k0 = 0; k0 < K; k0 += 32) {
    v16bf av0 = *(const v16bf*)(a0 + k0);
    v16bf av1 = *(const v16bf*)(a1 + k0);
#pragma unroll
    for (int t = 0; t < 4; ++t) {
      v16bf bv = *(const v16bf*)(b0 + (size_t)t * 16 * K + k0);
      c[0][t] = WMMA_BF16(av0, bv, c[0][t]);
      c[1][t] = WMMA_BF16(av1, bv, c[1][t]);
    }
  }

#pragma unroll
  for (int mi = 0; mi < 2; ++mi) {
#pragma unroll
    for (int t = 0; t < 4; ++t) {
      int n = n0 + t * 16 + nl;
#pragma unroll
      for (int j = 0; j < 8; ++j) {
        int m = m0 + mi * 16 + half * 8 + j;
        float val = c[mi][t][j] * cscale;
        if constexpr (MODE == 2) {
          ((float*)Out)[(size_t)m * N + n] = val;
        } else {
          int h  = n >> 6;          // head
          int dh = n & 63;          // within-head dim
          int bi = m >> 11;         // batch   (S=2048)
          int s  = m & (SS - 1);    // seq pos
          __bf16 v = (__bf16)val;
          if constexpr (MODE == 0)
            ((__bf16*)Out)[(((size_t)(bi * HH + h) * SS + s) << 6) + dh] = v;
          else
            ((__bf16*)Out)[((size_t)(bi * HH + h) * DHH + dh) * SS + s] = v;
        }
      }
    }
  }
}

// ---------------- flash attention (transposed formulation) ---------------
// Scores computed as S' = K_chunk @ Q^T (Q pre-scaled by 1/sqrt(DH) in the
// projection GEMM) so each lane owns one QUERY column: softmax reduction is
// in-lane + one shfl_xor(16). Output accumulated as O' = V^T @ P^T so the
// per-query alpha / 1/l rescales stay per-lane. The masked boundary chunk is
// peeled out of the hot loop (MASKED=false path has zero compares/selects).
template <bool MASKED>
__device__ __forceinline__ void attn_chunk(
    int j0, int valid, int nl, int half, int koff,
    const __bf16* __restrict__ Kb, const __bf16* __restrict__ Vb,
    __bf16* __restrict__ myp, v16bf qb0, v16bf qb1,
    v8f& o0, v8f& o1, v8f& o2, v8f& o3, float& mi, float& li) {
  v8f zero = {};

  // ---- S'[32 keys x 16 queries] = K_chunk @ Q^T  (2 C-fragments) ----
  v16bf ka;
  ka = *(const v16bf*)(Kb + (size_t)(j0 + nl) * DHH + koff);
  v8f sa = WMMA_BF16(ka, qb0, zero);
  ka = *(const v16bf*)(Kb + (size_t)(j0 + nl) * DHH + 32 + koff);
  sa = WMMA_BF16(ka, qb1, sa);
  ka = *(const v16bf*)(Kb + (size_t)(j0 + 16 + nl) * DHH + koff);
  v8f sb = WMMA_BF16(ka, qb0, zero);
  ka = *(const v16bf*)(Kb + (size_t)(j0 + 16 + nl) * DHH + 32 + koff);
  sb = WMMA_BF16(ka, qb1, sb);

  // ---- mask only in the peeled boundary chunk ----
  float sv[16];
#pragma unroll
  for (int j = 0; j < 8; ++j) {
    if constexpr (MASKED) {
      int kA = j0 + j + 8 * half;
      sv[j]     = (kA      < valid) ? sa[j] : -1e6f;
      sv[8 + j] = (kA + 16 < valid) ? sb[j] : -1e6f;
    } else {
      sv[j]     = sa[j];
      sv[8 + j] = sb[j];
    }
  }

  // ---- online softmax: in-lane reduce over 16 keys + 1 shuffle ----
  float rmax = sv[0];
#pragma unroll
  for (int i = 1; i < 16; ++i) rmax = fmaxf(rmax, sv[i]);
  rmax = fmaxf(rmax, __shfl_xor(rmax, 16, 32));
  float mnew  = fmaxf(mi, rmax);
  float alpha = __expf(mi - mnew);
  mi = mnew;

  float e[16];
  float rsum = 0.0f;
#pragma unroll
  for (int i = 0; i < 16; ++i) { e[i] = __expf(sv[i] - mnew); rsum += e[i]; }
  rsum += __shfl_xor(rsum, 16, 32);
  li = li * alpha + rsum;

#pragma unroll
  for (int j = 0; j < 8; ++j) {
    o0[j] *= alpha; o1[j] *= alpha; o2[j] *= alpha; o3[j] *= alpha;
  }

  // ---- P -> LDS as [query][key] row-major ----
#pragma unroll
  for (int j = 0; j < 8; ++j) {
    myp[nl * 32 + j + 8 * half]      = (__bf16)e[j];
    myp[nl * 32 + 16 + j + 8 * half] = (__bf16)e[8 + j];
  }

  // ---- O'[64 dh x 16 q] += V^T_chunk @ P^T (4 WMMAs) ----
  v16bf pb = *(const v16bf*)(myp + nl * 32 + koff);   // B frag: lane=query col
  v16bf va;
  va = *(const v16bf*)(Vb + (size_t)(0  + nl) * SS + j0 + koff);
  o0 = WMMA_BF16(va, pb, o0);
  va = *(const v16bf*)(Vb + (size_t)(16 + nl) * SS + j0 + koff);
  o1 = WMMA_BF16(va, pb, o1);
  va = *(const v16bf*)(Vb + (size_t)(32 + nl) * SS + j0 + koff);
  o2 = WMMA_BF16(va, pb, o2);
  va = *(const v16bf*)(Vb + (size_t)(48 + nl) * SS + j0 + koff);
  o3 = WMMA_BF16(va, pb, o3);
}

// Q,K: [B,H,S,DH] bf16 ; Vt: [B,H,DH,S] bf16 ; out: bf16 [B,S,D]
__global__ void attn_kernel(const __bf16* __restrict__ Q,
                            const __bf16* __restrict__ Kh,
                            const __bf16* __restrict__ Vt,
                            const int* __restrict__ valid_lens,
                            __bf16* __restrict__ Outb) {
  __shared__ __align__(128) __bf16 pbuf[8][16 * 32];   // per-wave P[query][key]

  int wave = threadIdx.x >> 5;
  int lane = threadIdx.x & 31;
  int w  = blockIdx.x * 8 + wave;          // 8192 waves total
  int qt = w & 127;                        // 128 query tiles
  int bh = w >> 7;
  int h  = bh & (HH - 1);
  int b  = bh >> 4;
  int q0 = qt * 16;

  const __bf16* Qb = Q  + (size_t)(b * HH + h) * SS * DHH;
  const __bf16* Kb = Kh + (size_t)(b * HH + h) * SS * DHH;
  const __bf16* Vb = Vt + (size_t)(b * HH + h) * DHH * SS;
  int valid = valid_lens[b];               // 1..S-1

  int nl = lane & 15, half = lane >> 4;
  int koff = half * 16;

  // Q as B-fragments (lane = query column), reused for all key chunks
  v16bf qb0 = *(const v16bf*)(Qb + (size_t)(q0 + nl) * DHH + koff);        // dh 0..31
  v16bf qb1 = *(const v16bf*)(Qb + (size_t)(q0 + nl) * DHH + 32 + koff);   // dh 32..63

  v8f o0 = {}, o1 = {}, o2 = {}, o3 = {};
  float mi = -1e30f, li = 0.0f;            // per-lane = per-query state
  __bf16* myp = &pbuf[wave][0];

  // ---- hot loop: fully-valid chunks, no masking at all ----
  int valid_full = valid & ~31;
  for (int j0 = 0; j0 < valid_full; j0 += 32) {
    // prefetch next chunk's K/V rows (gfx1250 global_prefetch_b8); rows stay
    // in-range because valid <= S-1.
    __builtin_prefetch(Kb + (size_t)(j0 + 32 + nl) * DHH, 0, 3);
    __builtin_prefetch(Vb + (size_t)nl * SS + j0 + 32, 0, 3);
    attn_chunk<false>(j0, valid, nl, half, koff, Kb, Vb, myp,
                      qb0, qb1, o0, o1, o2, o3, mi, li);
  }
  // ---- peeled boundary chunk with per-key masking ----
  if (valid_full < valid) {
    attn_chunk<true>(valid_full, valid, nl, half, koff, Kb, Vb, myp,
                     qb0, qb1, o0, o1, o2, o3, mi, li);
  }

  // ---- normalize (per-lane scalar) and store bf16 [B,S,D] ----
  // O' frag t: vgpr j -> dh = t*16 + j + 8*half, lane -> query = nl.
  // For a fixed lane the 8 dh values are consecutive -> one 16B store each.
  float inv = 1.0f / li;
  size_t rowbase = ((size_t)(b * SS + q0 + nl)) * DD + h * DHH + 8 * half;
  v8f oacc[4] = {o0, o1, o2, o3};
#pragma unroll
  for (int t = 0; t < 4; ++t) {
    v8bf pk;
#pragma unroll
    for (int j = 0; j < 8; ++j) pk[j] = (__bf16)(oacc[t][j] * inv);
    *(v8bf*)(Outb + rowbase + t * 16) = pk;
  }
}

// ---------------- launcher ----------------
extern "C" void kernel_launch(void* const* d_in, const int* in_sizes, int n_in,
                              void* d_out, int out_size, void* d_ws, size_t ws_size,
                              hipStream_t stream) {
  (void)in_sizes; (void)n_in; (void)out_size; (void)ws_size;

  const float* queries   = (const float*)d_in[0];
  const float* keys      = (const float*)d_in[1];
  const float* values    = (const float*)d_in[2];
  const int*   valid_len = (const int*)  d_in[3];
  const float* W_q       = (const float*)d_in[4];
  const float* W_k       = (const float*)d_in[5];
  const float* W_v       = (const float*)d_in[6];
  const float* W_o       = (const float*)d_in[7];
  float* out             = (float*)d_out;

  const size_t XBYTES = (size_t)BB * SS * DD * sizeof(__bf16); // 16 MB
  const size_t WBYTES = (size_t)DD * DD * sizeof(__bf16);      //  2 MB

  char* ws = (char*)d_ws;
  __bf16* Xq  = (__bf16*)(ws);
  __bf16* Xk  = (__bf16*)(ws + XBYTES);
  __bf16* Xv  = (__bf16*)(ws + 2 * XBYTES);
  __bf16* WqT = (__bf16*)(ws + 3 * XBYTES);
  __bf16* WkT = (__bf16*)(ws + 3 * XBYTES + WBYTES);
  __bf16* WvT = (__bf16*)(ws + 3 * XBYTES + 2 * WBYTES);
  __bf16* WoT = (__bf16*)(ws + 3 * XBYTES + 3 * WBYTES);
  __bf16* Qh  = (__bf16*)(ws + 3 * XBYTES + 4 * WBYTES);
  __bf16* Kh  = (__bf16*)(ws + 4 * XBYTES + 4 * WBYTES);
  __bf16* Vt  = (__bf16*)(ws + 5 * XBYTES + 4 * WBYTES);
  __bf16* AO  = (__bf16*)(ws + 6 * XBYTES + 4 * WBYTES);

  const int nX = BB * SS * DD;                 // 8388608
  dim3 blk(256);

  // 1) convert activations to bf16
  cvt_bf16_kernel<<<nX / (256 * 8), blk, 0, stream>>>(queries, Xq, nX);
  cvt_bf16_kernel<<<nX / (256 * 8), blk, 0, stream>>>(keys,    Xk, nX);
  cvt_bf16_kernel<<<nX / (256 * 8), blk, 0, stream>>>(values,  Xv, nX);

  // 2) transpose + convert weights
  const int nW = DD * DD;                      // 1048576
  transpose_w_kernel<<<nW / 256, blk, 0, stream>>>(W_q, WqT);
  transpose_w_kernel<<<nW / 256, blk, 0, stream>>>(W_k, WkT);
  transpose_w_kernel<<<nW / 256, blk, 0, stream>>>(W_v, WvT);
  transpose_w_kernel<<<nW / 256, blk, 0, stream>>>(W_o, WoT);

  // 3) projections (WMMA, 32x64 per wave): Q (pre-scaled by 1/sqrt(DH)),
  //    K head-split; V head-split-transposed
  dim3 ggrid(DD / 256, (BB * SS) / 64);        // (4, 128)
  gemm_bf16_kernel<0><<<ggrid, blk, 0, stream>>>(Xq, WqT, (void*)Qh, 0.125f);
  gemm_bf16_kernel<0><<<ggrid, blk, 0, stream>>>(Xk, WkT, (void*)Kh, 1.0f);
  gemm_bf16_kernel<1><<<ggrid, blk, 0, stream>>>(Xv, WvT, (void*)Vt, 1.0f);

  // 4) flash attention (WMMA + transposed online softmax, peeled masking)
  int nWaves = BB * HH * (SS / 16);            // 8192
  attn_kernel<<<nWaves / 8, blk, 0, stream>>>(Qh, Kh, Vt, valid_len, AO);

  // 5) output projection (WMMA) -> fp32 d_out
  gemm_bf16_kernel<2><<<ggrid, blk, 0, stream>>>(AO, WoT, (void*)out, 1.0f);
}
